// MoELayer_35253091566176
// MI455X (gfx1250) — matrix operand order repaired and checked by reading, not compiled, not run on hardware
//
#include <hip/hip_runtime.h>
#include <math.h>

// ---------------------------------------------------------------------------
// MoE FFN for MI455X (gfx1250): bf16 WMMA (v_wmma_f32_16x16x32_bf16) with
// token gathering (top-2 routing => ~4x fewer FLOPs than dense reference).
// This revision: 2-stage software pipeline on the WMMA fragment loads so the
// global loads for k+1 overlap the WMMAs for k (break s_wait_loadcnt 0 chain).
// ---------------------------------------------------------------------------

typedef __bf16 bf16_t;
typedef __attribute__((ext_vector_type(16))) __bf16 v16bf;
typedef __attribute__((ext_vector_type(8)))  __bf16 v8bf;
typedef __attribute__((ext_vector_type(4)))  __bf16 v4bf;
typedef __attribute__((ext_vector_type(8)))  float  v8f;

#define TT 8192          // tokens (B*S)
#define DD 1024          // model dim
#define FF 4096          // ffn dim
#define EE 8             // experts
#define F2 (2*FF)

union FragAB { v16bf v; v8bf h[2]; };

#define WMMA_BF16(a, b, c) \
    __builtin_amdgcn_wmma_f32_16x16x32_bf16(false, (a), false, (b), (short)0, (c), false, false)

// ---------------------------------------------------------------- utilities
__global__ void zero_i32_kernel(int* __restrict__ p, int n) {
    int i = blockIdx.x * blockDim.x + threadIdx.x;
    if (i < n) p[i] = 0;
}

__global__ void zero_f4_kernel(float4* __restrict__ p, int n4) {
    int i = blockIdx.x * blockDim.x + threadIdx.x;
    if (i < n4) p[i] = make_float4(0.f, 0.f, 0.f, 0.f);
}

__global__ void cvt_x_kernel(const float4* __restrict__ in, v4bf* __restrict__ outp, int n4) {
    int i = blockIdx.x * blockDim.x + threadIdx.x;
    if (i >= n4) return;
    float4 v = in[i];
    v4bf o;
    o[0] = (bf16_t)v.x; o[1] = (bf16_t)v.y; o[2] = (bf16_t)v.z; o[3] = (bf16_t)v.w;
    outp[i] = o;
}

// Batched transpose + fp32->bf16: in[b][R][C] -> out[b][C][R].  Block (32,8).
__global__ __launch_bounds__(256)
void transpose_cvt_kernel(const float* __restrict__ in, bf16_t* __restrict__ outp,
                          int R, int C) {
    __shared__ float tile[32][33];
    const int b  = blockIdx.z;
    const float* inb = in + (size_t)b * R * C;
    bf16_t*      ob  = outp + (size_t)b * R * C;
    const int c0 = blockIdx.x * 32, r0 = blockIdx.y * 32;
    #pragma unroll
    for (int i = 0; i < 4; ++i) {
        int r = r0 + threadIdx.y + i * 8;
        tile[threadIdx.y + i * 8][threadIdx.x] = inb[(size_t)r * C + c0 + threadIdx.x];
    }
    __syncthreads();
    #pragma unroll
    for (int i = 0; i < 4; ++i) {
        int c = c0 + threadIdx.y + i * 8;
        ob[(size_t)c * R + r0 + threadIdx.x] = (bf16_t)tile[threadIdx.x][threadIdx.y + i * 8];
    }
}

// -------------------------------------------------------------- gate/router
// One wave32 per token: lane-parallel dot over D, shuffle-reduce, softmax,
// top-2, combine weights, per-expert token list via atomic slot counters.
__global__ __launch_bounds__(256)
void gate_kernel(const float* __restrict__ x, const float* __restrict__ gw,
                 const float* __restrict__ gb, float* __restrict__ combine,
                 float* __restrict__ probs, int* __restrict__ list,
                 int* __restrict__ cnt) {
    const int lane = threadIdx.x & 31;
    const int wave = threadIdx.x >> 5;
    const int t = blockIdx.x * 8 + wave;

    float acc[EE];
    #pragma unroll
    for (int e = 0; e < EE; ++e) acc[e] = 0.f;

    const float* xr = x + (size_t)t * DD;
    for (int d = lane; d < DD; d += 32) {
        float xv = xr[d];
        #pragma unroll
        for (int e = 0; e < EE; ++e) acc[e] += xv * gw[d * EE + e];
    }
    #pragma unroll
    for (int e = 0; e < EE; ++e) {
        float v = acc[e];
        for (int off = 16; off > 0; off >>= 1) v += __shfl_down(v, off);
        acc[e] = v;
    }
    if (lane == 0) {
        float mx = -3.4e38f;
        #pragma unroll
        for (int e = 0; e < EE; ++e) { acc[e] += gb[e]; mx = fmaxf(mx, acc[e]); }
        float s = 0.f;
        #pragma unroll
        for (int e = 0; e < EE; ++e) { acc[e] = __expf(acc[e] - mx); s += acc[e]; }
        float inv = 1.f / s;
        #pragma unroll
        for (int e = 0; e < EE; ++e) {
            acc[e] *= inv;
            probs[(size_t)t * EE + e] = acc[e];
        }
        int i1 = 0;
        #pragma unroll
        for (int e = 1; e < EE; ++e) if (acc[e] > acc[i1]) i1 = e;
        int i2 = (i1 == 0) ? 1 : 0;
        #pragma unroll
        for (int e = 0; e < EE; ++e) if (e != i1 && acc[e] > acc[i2]) i2 = e;
        float denom = acc[i1] + acc[i2] + 1e-6f;
        combine[(size_t)t * EE + i1] = acc[i1] / denom;
        combine[(size_t)t * EE + i2] = acc[i2] / denom;
        int s1 = atomicAdd(&cnt[i1], 1); list[i1 * TT + s1] = t;
        int s2 = atomicAdd(&cnt[i2], 1); list[i2 * TT + s2] = t;
    }
}

// Deterministic load-balance loss reduction (fixed-order tree reduce).
__global__ __launch_bounds__(256)
void lb_kernel(const float* __restrict__ probs, const int* __restrict__ cnt,
               float* __restrict__ out_lb) {
    __shared__ float sh[256];
    __shared__ float esum[EE];
    for (int e = 0; e < EE; ++e) {
        float p = 0.f;
        for (int t = threadIdx.x; t < TT; t += 256) p += probs[(size_t)t * EE + e];
        sh[threadIdx.x] = p;
        __syncthreads();
        for (int s = 128; s > 0; s >>= 1) {
            if ((int)threadIdx.x < s) sh[threadIdx.x] += sh[threadIdx.x + s];
            __syncthreads();
        }
        if (threadIdx.x == 0) esum[e] = sh[0];
        __syncthreads();
    }
    if (threadIdx.x == 0) {
        float lb = 0.f;
        for (int e = 0; e < EE; ++e)
            lb += ((float)cnt[e] / (float)TT) * (esum[e] / (float)TT);
        *out_lb = 0.01f * (float)EE * lb;
    }
}

// ------------------------------------------------------------- expert GEMMs
// fc1 (gathered): act[slot, 0:F] = a*gelu(g),  h = x[list[slot]] @ fc1_w[e] + b
// Block = 8 waves, tile 128(slots) x 64(F cols); each wave: 2x2 WMMA tiles
// for both the 'a' and 'g' halves.  Fragment loads double-buffered.
__global__ __launch_bounds__(256)
void fc1_kernel(const bf16_t* __restrict__ xbf, const bf16_t* __restrict__ w1t,
                const float* __restrict__ fc1_b, const int* __restrict__ list,
                const int* __restrict__ cnt, bf16_t* __restrict__ act, int e) {
    const int count = cnt[e];
    const int m0 = blockIdx.y * 128;
    if (m0 >= count) return;
    const int n0    = blockIdx.x * 64;
    const int lane  = threadIdx.x & 31;
    const int wave  = threadIdx.x >> 5;
    const int wm    = wave >> 1;          // 0..3 -> M sub-tile of 32
    const int wn    = wave & 1;           // 0..1 -> N sub-tile of 32
    const int l15   = lane & 15;
    const int lhalf = lane >> 4;

    const bf16_t* w1e   = w1t + (size_t)e * F2 * DD;
    const int*    liste = list + e * TT;

    // gathered A rows (clamped for tail padding; results there are discarded)
    int row0 = m0 + wm * 32 + l15;
    int row1 = row0 + 16;
    int t0 = liste[row0 < count ? row0 : count - 1];
    int t1 = liste[row1 < count ? row1 : count - 1];
    const bf16_t* arow0 = xbf + (size_t)t0 * DD;
    const bf16_t* arow1 = xbf + (size_t)t1 * DD;

    // B columns (transposed weights: w1t[col][k], K contiguous)
    int colA = n0 + wn * 32 + l15;
    const bf16_t* bA0 = w1e + (size_t)colA * DD;
    const bf16_t* bA1 = bA0 + (size_t)16 * DD;
    const bf16_t* bG0 = w1e + (size_t)(FF + colA) * DD;
    const bf16_t* bG1 = bG0 + (size_t)16 * DD;

    v8f accA[2][2] = {};
    v8f accG[2][2] = {};

    // double-buffered fragments: loads for step k+1 overlap WMMAs for step k
    FragAB a0[2], a1[2], fA0[2], fA1[2], fG0[2], fG1[2];

    auto load_frags = [&](int kb, int s) {
        const int ka = kb + lhalf * 16;          // A layout: 16 contiguous K
        a0[s].h[0] = *(const v8bf*)(arow0 + ka);
        a0[s].h[1] = *(const v8bf*)(arow0 + ka + 8);
        a1[s].h[0] = *(const v8bf*)(arow1 + ka);
        a1[s].h[1] = *(const v8bf*)(arow1 + ka + 8);
        const int kv = kb + lhalf * 16;          // B layout: 16 contiguous K
        fA0[s].h[0] = *(const v8bf*)(bA0 + kv);  fA0[s].h[1] = *(const v8bf*)(bA0 + kv + 8);
        fA1[s].h[0] = *(const v8bf*)(bA1 + kv);  fA1[s].h[1] = *(const v8bf*)(bA1 + kv + 8);
        fG0[s].h[0] = *(const v8bf*)(bG0 + kv);  fG0[s].h[1] = *(const v8bf*)(bG0 + kv + 8);
        fG1[s].h[0] = *(const v8bf*)(bG1 + kv);  fG1[s].h[1] = *(const v8bf*)(bG1 + kv + 8);
    };

    load_frags(0, 0);
    #pragma unroll 4
    for (int kb = 0; kb < DD; kb += 32) {
        const int cur = (kb >> 5) & 1;
        const int nxt = cur ^ 1;
        if (kb + 32 < DD) load_frags(kb + 32, nxt);

        accA[0][0] = WMMA_BF16(a0[cur].v, fA0[cur].v, accA[0][0]);
        accA[0][1] = WMMA_BF16(a0[cur].v, fA1[cur].v, accA[0][1]);
        accA[1][0] = WMMA_BF16(a1[cur].v, fA0[cur].v, accA[1][0]);
        accA[1][1] = WMMA_BF16(a1[cur].v, fA1[cur].v, accA[1][1]);
        accG[0][0] = WMMA_BF16(a0[cur].v, fG0[cur].v, accG[0][0]);
        accG[0][1] = WMMA_BF16(a0[cur].v, fG1[cur].v, accG[0][1]);
        accG[1][0] = WMMA_BF16(a1[cur].v, fG0[cur].v, accG[1][0]);
        accG[1][1] = WMMA_BF16(a1[cur].v, fG1[cur].v, accG[1][1]);
    }

    // epilogue: bias + SwiGLU (exact gelu) -> bf16 activation buffer
    #pragma unroll
    for (int mi = 0; mi < 2; ++mi) {
        int mbase = m0 + wm * 32 + mi * 16 + lhalf * 8;
        #pragma unroll
        for (int nj = 0; nj < 2; ++nj) {
            int col = n0 + wn * 32 + nj * 16 + l15;
            float ba = fc1_b[(size_t)e * F2 + col];
            float bg = fc1_b[(size_t)e * F2 + FF + col];
            #pragma unroll
            for (int r = 0; r < 8; ++r) {
                float av = accA[mi][nj][r] + ba;
                float gv = accG[mi][nj][r] + bg;
                float gel = 0.5f * gv * (1.0f + erff(gv * 0.70710678118654752f));
                act[(size_t)(mbase + r) * FF + col] = (bf16_t)(av * gel);
            }
        }
    }
}

// fc2 (scatter): out[token] += combine[token,e] * (act[slot] @ fc2_w[e] + b)
__global__ __launch_bounds__(256)
void fc2_kernel(const bf16_t* __restrict__ act, const bf16_t* __restrict__ w2t,
                const float* __restrict__ fc2_b, const int* __restrict__ list,
                const int* __restrict__ cnt, const float* __restrict__ combine,
                float* __restrict__ out, int e) {
    const int count = cnt[e];
    const int m0 = blockIdx.y * 128;
    if (m0 >= count) return;
    const int n0    = blockIdx.x * 64;
    const int lane  = threadIdx.x & 31;
    const int wave  = threadIdx.x >> 5;
    const int wm    = wave >> 1;
    const int wn    = wave & 1;
    const int l15   = lane & 15;
    const int lhalf = lane >> 4;

    const int* liste = list + e * TT;

    int row0 = m0 + wm * 32 + l15;
    const bf16_t* arow0 = act + (size_t)row0 * FF;
    const bf16_t* arow1 = arow0 + (size_t)16 * FF;

    int col = n0 + wn * 32 + l15;
    const bf16_t* b0 = w2t + (size_t)e * DD * FF + (size_t)col * FF;
    const bf16_t* b1 = b0 + (size_t)16 * FF;

    v8f acc[2][2] = {};

    FragAB a0[2], a1[2], f0[2], f1[2];
    auto load_frags = [&](int kb, int s) {
        const int ka = kb + lhalf * 16;
        a0[s].h[0] = *(const v8bf*)(arow0 + ka);
        a0[s].h[1] = *(const v8bf*)(arow0 + ka + 8);
        a1[s].h[0] = *(const v8bf*)(arow1 + ka);
        a1[s].h[1] = *(const v8bf*)(arow1 + ka + 8);
        const int kv = kb + lhalf * 16;
        f0[s].h[0] = *(const v8bf*)(b0 + kv);  f0[s].h[1] = *(const v8bf*)(b0 + kv + 8);
        f1[s].h[0] = *(const v8bf*)(b1 + kv);  f1[s].h[1] = *(const v8bf*)(b1 + kv + 8);
    };

    load_frags(0, 0);
    #pragma unroll 4
    for (int kb = 0; kb < FF; kb += 32) {
        const int cur = (kb >> 5) & 1;
        const int nxt = cur ^ 1;
        if (kb + 32 < FF) load_frags(kb + 32, nxt);

        acc[0][0] = WMMA_BF16(a0[cur].v, f0[cur].v, acc[0][0]);
        acc[0][1] = WMMA_BF16(a0[cur].v, f1[cur].v, acc[0][1]);
        acc[1][0] = WMMA_BF16(a1[cur].v, f0[cur].v, acc[1][0]);
        acc[1][1] = WMMA_BF16(a1[cur].v, f1[cur].v, acc[1][1]);
    }

    #pragma unroll
    for (int mi = 0; mi < 2; ++mi) {
        int mbase = m0 + wm * 32 + mi * 16 + lhalf * 8;
        #pragma unroll
        for (int nj = 0; nj < 2; ++nj) {
            int c = n0 + wn * 32 + nj * 16 + l15;
            float bb = fc2_b[(size_t)e * DD + c];
            #pragma unroll
            for (int r = 0; r < 8; ++r) {
                int m = mbase + r;
                if (m < count) {
                    int t = liste[m];
                    float w = combine[(size_t)t * EE + e];
                    out[(size_t)t * DD + c] += w * (acc[mi][nj][r] + bb);
                }
            }
        }
    }
}

// ------------------------------------------------------------------- launch
extern "C" void kernel_launch(void* const* d_in, const int* in_sizes, int n_in,
                              void* d_out, int out_size, void* d_ws, size_t ws_size,
                              hipStream_t stream) {
    (void)in_sizes; (void)n_in; (void)out_size; (void)ws_size;
    const float* x      = (const float*)d_in[0];
    const float* gate_w = (const float*)d_in[1];
    const float* gate_b = (const float*)d_in[2];
    const float* fc1_w  = (const float*)d_in[3];
    const float* fc1_b  = (const float*)d_in[4];
    const float* fc2_w  = (const float*)d_in[5];
    const float* fc2_b  = (const float*)d_in[6];
    float* out = (float*)d_out;

    // workspace layout (~273 MiB)
    char* wsb = (char*)d_ws;
    size_t off = 0;
    bf16_t* xbf = (bf16_t*)(wsb + off); off += (size_t)TT * DD * 2;       // 16 MiB
    bf16_t* w1t = (bf16_t*)(wsb + off); off += (size_t)EE * F2 * DD * 2;  // 128 MiB
    bf16_t* w2t = (bf16_t*)(wsb + off); off += (size_t)EE * DD * FF * 2;  // 64 MiB
    bf16_t* act = (bf16_t*)(wsb + off); off += (size_t)TT * FF * 2;       // 64 MiB
    float* combine = (float*)(wsb + off); off += (size_t)TT * EE * 4;
    float* probs   = (float*)(wsb + off); off += (size_t)TT * EE * 4;
    int*   list    = (int*)(wsb + off);   off += (size_t)EE * TT * 4;
    int*   cnt     = (int*)(wsb + off);   off += 256;

    const int n4 = (TT * DD) / 4;

    zero_i32_kernel<<<1, 64, 0, stream>>>(cnt, 64);
    cvt_x_kernel<<<(n4 + 255) / 256, 256, 0, stream>>>((const float4*)x, (v4bf*)xbf, n4);
    transpose_cvt_kernel<<<dim3(F2 / 32, DD / 32, EE), dim3(32, 8), 0, stream>>>(fc1_w, w1t, DD, F2);
    transpose_cvt_kernel<<<dim3(DD / 32, FF / 32, EE), dim3(32, 8), 0, stream>>>(fc2_w, w2t, FF, DD);
    gate_kernel<<<TT / 8, 256, 0, stream>>>(x, gate_w, gate_b, combine, probs, list, cnt);
    lb_kernel<<<1, 256, 0, stream>>>(probs, cnt, out + (size_t)TT * DD);
    zero_f4_kernel<<<(n4 + 255) / 256, 256, 0, stream>>>((float4*)out, n4);

    for (int e = 0; e < EE; ++e) {
        fc1_kernel<<<dim3(FF / 64, TT / 128), 256, 0, stream>>>(xbf, w1t, fc1_b, list, cnt, act, e);
        fc2_kernel<<<dim3(DD / 64, TT / 128), 256, 0, stream>>>(act, w2t, fc2_b, list, cnt, combine, out, e);
    }
}
